// GRULayer_58643483460152
// MI455X (gfx1250) — matrix operand order repaired and checked by reading, not compiled, not run on hardware
//
#include <hip/hip_runtime.h>

// ---------------------------------------------------------------------------
// GRU layer for gfx1250 (MI455X), wave32 + V_WMMA_F32_16X16X32_BF16.
// B=32, T=2048, HE=H=512.
// Phase 0a: transpose+convert 6 weight matrices to bf16 (wT[n][k]) in ws.
// Phase 0b: convert x to bf16 (row-major) in ws.
// Phase 1 : fused input projections via WMMA -> u_x, r_x (ws), c_x (d_out).
// Phase 2 : 2 persistent workgroups (16 independent batches each), h state in
//           LDS (fp32 truth + bf16 mirror), 6 WMMAs per A-fragment load.
// ---------------------------------------------------------------------------

typedef __attribute__((ext_vector_type(16))) __bf16 v16bf;
typedef __attribute__((ext_vector_type(8)))  float  v8f;

#define GRU_B  32
#define GRU_T  2048
#define GRU_H  512
#define GRU_BT (GRU_B * GRU_T)

// float -> bf16, round to nearest even (bit-exact, no reliance on cast lowering)
__device__ __forceinline__ __bf16 f2bf(float f) {
    union { float f; unsigned u; } v; v.f = f;
    unsigned r = v.u + 0x7FFFu + ((v.u >> 16) & 1u);
    union { unsigned short s; __bf16 b; } o; o.s = (unsigned short)(r >> 16);
    return o.b;
}

// A-fragment (16x32, MxK) from bf16 row-major [row][k], ld elems (global or LDS).
// ISA layout: lanes 0-15 = rows M, halves 0-7 -> K k0..k0+7, 8-15 -> k0+16..k0+23
//             lanes 16-31 same rows,          -> K k0+8..k0+15 / k0+24..k0+31
__device__ __forceinline__ v16bf frag_a_bf16(const __bf16* __restrict__ base, int ld,
                                             int row0, int k0, int lane) {
    const int half = (lane >> 4) & 1;
    const int r    = row0 + (lane & 15);
    const __bf16* p = base + (size_t)r * ld + k0 + half * 8;
    union U { uint4 u; __bf16 h[8]; };
    U a, b;
    a.u = *(const uint4*)(p);       // K k0+half*8    .. +7
    b.u = *(const uint4*)(p + 16);  // K k0+half*8+16 .. +7
    v16bf v;
#pragma unroll
    for (int j = 0; j < 8; ++j) { v[j] = a.h[j]; v[8 + j] = b.h[j]; }
    return v;
}

// B-fragment (32x16, KxN) from bf16 *transposed* weights wT[n][k], ld elems.
// Dense-B layout: lanes 0-15 = cols n, halves 0-7 -> K k0..k0+7, 8-15 -> k0+16..k0+23;
//                 lanes 16-31 same cols -> K k0+8..k0+15 / k0+24..k0+31.
__device__ __forceinline__ v16bf frag_b_bf16(const __bf16* __restrict__ wT, int ld,
                                             int n0, int k0, int lane) {
    const int n  = n0 + (lane & 15);
    const int ks = k0 + ((lane >> 4) & 1) * 8;
    const __bf16* p = wT + (size_t)n * ld + ks;
    union U { uint4 u; __bf16 h[8]; };
    U a, b;
    a.u = *(const uint4*)(p);
    b.u = *(const uint4*)(p + 16);
    v16bf v;
#pragma unroll
    for (int j = 0; j < 8; ++j) { v[j] = a.h[j]; v[8 + j] = b.h[j]; }
    return v;
}

#define WMMA_BF16(A, Bf, C) \
    __builtin_amdgcn_wmma_f32_16x16x32_bf16(false, (A), false, (Bf), (short)0, (C), false, false)

// ------------------------ Phase 0a: weight transpose ------------------------
__global__ __launch_bounds__(256) void wt_convert(const float* __restrict__ src,
                                                  __bf16* __restrict__ dst) {
    const int idx = blockIdx.x * 256 + threadIdx.x;   // 0 .. 512*512-1
    const int n = idx >> 9;
    const int k = idx & 511;
    dst[(size_t)n * GRU_H + k] = f2bf(src[(size_t)k * GRU_H + n]);
}

// ------------------------ Phase 0b: x fp32 -> bf16 --------------------------
// Each thread converts 8 consecutive floats (2x float4 in, 1x uint4 out).
__global__ __launch_bounds__(256) void x_convert(const float* __restrict__ src,
                                                 __bf16* __restrict__ dst) {
    const size_t i = (size_t)blockIdx.x * 256 + threadIdx.x;   // 8-elem chunk id
    const float4 a = ((const float4*)src)[2 * i];
    const float4 b = ((const float4*)src)[2 * i + 1];
    union { uint4 u; __bf16 h[8]; } o;
    o.h[0] = f2bf(a.x); o.h[1] = f2bf(a.y); o.h[2] = f2bf(a.z); o.h[3] = f2bf(a.w);
    o.h[4] = f2bf(b.x); o.h[5] = f2bf(b.y); o.h[6] = f2bf(b.z); o.h[7] = f2bf(b.w);
    ((uint4*)dst)[i] = o.u;
}

// --------------------------- Phase 1: projections ---------------------------
__global__ __launch_bounds__(256) void proj_kernel(
    const __bf16* __restrict__ xb,
    const __bf16* __restrict__ wT_ui, const __bf16* __restrict__ wT_ri,
    const __bf16* __restrict__ wT_ci,
    const float* __restrict__ bu, const float* __restrict__ br,
    const float* __restrict__ bc,
    float* __restrict__ u_x, float* __restrict__ r_x, float* __restrict__ c_x) {
    const int lane   = threadIdx.x & 31;
    const int gw     = blockIdx.x * 8 + (threadIdx.x >> 5);
    const int n_tile = gw & 31;           // 32 n-tiles over H
    const int m_tile = gw >> 5;           // 4096 m-tiles over B*T
    const int m0 = m_tile * 16, n0 = n_tile * 16;

    v8f acc_u = {}, acc_r = {}, acc_c = {};
    for (int k0 = 0; k0 < GRU_H; k0 += 32) {
        v16bf a  = frag_a_bf16(xb, GRU_H, m0, k0, lane);
        v16bf b0 = frag_b_bf16(wT_ui, GRU_H, n0, k0, lane);
        v16bf b1 = frag_b_bf16(wT_ri, GRU_H, n0, k0, lane);
        v16bf b2 = frag_b_bf16(wT_ci, GRU_H, n0, k0, lane);
        acc_u = WMMA_BF16(a, b0, acc_u);
        acc_r = WMMA_BF16(a, b1, acc_r);
        acc_c = WMMA_BF16(a, b2, acc_c);
    }

    const int col   = n0 + (lane & 15);
    const int rbase = m0 + ((lane >> 4) & 1) * 8;
    const float vbu = bu[col], vbr = br[col], vbc = bc[col];
#pragma unroll
    for (int j = 0; j < 8; ++j) {
        const size_t idx = (size_t)(rbase + j) * GRU_H + col;
        u_x[idx] = acc_u[j] + vbu;
        r_x[idx] = acc_r[j] + vbr;
        c_x[idx] = acc_c[j] + vbc;
    }
}

// --------------------------- Phase 2: recurrence ----------------------------
// 2 workgroups x 512 threads. Each WG owns 16 independent batches (no cross-WG
// sync needed). h state lives in LDS. k-outer loop: each A-fragment (LDS h)
// feeds 6 WMMAs (2 n-tiles x 3 gates).
__global__ __launch_bounds__(512) void scan_kernel(
    const __bf16* __restrict__ wT_uh, const __bf16* __restrict__ wT_rh,
    const __bf16* __restrict__ wT_ch,
    const float* __restrict__ u_x, const float* __restrict__ r_x,
    float* __restrict__ out /* holds c_x on entry per-(b,t) slot */) {
    __shared__ float  h_f[16 * GRU_H];   // 32 KB fp32 hidden state
    __shared__ __bf16 h_b[16 * GRU_H];   // 16 KB bf16 mirror

    const int tid   = threadIdx.x;
    const int lane  = tid & 31;
    const int wave  = tid >> 5;           // 0..15, owns n-tiles {2w, 2w+1}
    const int b0    = blockIdx.x * 16;    // global batch base (0 or 16)
    const int group = (lane >> 4) & 1;

    for (int i = tid; i < 16 * GRU_H; i += 512) {
        h_f[i] = 0.0f;
        h_b[i] = f2bf(0.0f);
    }
    __syncthreads();

    for (int t = 0; t < GRU_T; ++t) {
        v8f au[2] = {{}, {}}, ar[2] = {{}, {}}, ac[2] = {{}, {}};
        for (int k0 = 0; k0 < GRU_H; k0 += 32) {
            const v16bf a = frag_a_bf16(h_b, GRU_H, 0, k0, lane);  // shared by 6 WMMAs
#pragma unroll
            for (int nt = 0; nt < 2; ++nt) {
                const int n0 = (wave * 2 + nt) * 16;
                v16bf fbu = frag_b_bf16(wT_uh, GRU_H, n0, k0, lane);
                v16bf fbr = frag_b_bf16(wT_rh, GRU_H, n0, k0, lane);
                v16bf fbc = frag_b_bf16(wT_ch, GRU_H, n0, k0, lane);
                au[nt] = WMMA_BF16(a, fbu, au[nt]);
                ar[nt] = WMMA_BF16(a, fbr, ar[nt]);
                ac[nt] = WMMA_BF16(a, fbc, ac[nt]);
            }
        }

        float hn[2][8];
#pragma unroll
        for (int nt = 0; nt < 2; ++nt) {
            const int col = (wave * 2 + nt) * 16 + (lane & 15);
#pragma unroll
            for (int j = 0; j < 8; ++j) {
                const int lb = group * 8 + j;                         // local batch
                const size_t idx = ((size_t)(b0 + lb) * GRU_T + t) * GRU_H + col;
                const float ux = u_x[idx];
                const float rx = r_x[idx];
                const float cx = out[idx];                            // staged c_x
                const float hp = h_f[lb * GRU_H + col];
                const float u  = 1.0f / (1.0f + __expf(-(ux + au[nt][j])));
                const float r  = 1.0f / (1.0f + __expf(-(rx + ar[nt][j])));
                const float c  = tanhf(cx + r * ac[nt][j]);
                const float h  = (1.0f - u) * hp + u * c;
                out[idx]  = h;
                hn[nt][j] = h;
            }
        }
        __syncthreads();   // all waves done reading h for step t
#pragma unroll
        for (int nt = 0; nt < 2; ++nt) {
            const int col = (wave * 2 + nt) * 16 + (lane & 15);
#pragma unroll
            for (int j = 0; j < 8; ++j) {
                const int lb = group * 8 + j;
                h_f[lb * GRU_H + col] = hn[nt][j];
                h_b[lb * GRU_H + col] = f2bf(hn[nt][j]);
            }
        }
        __syncthreads();   // new h visible to all waves
    }
}

// ------------------------------- launcher -----------------------------------
extern "C" void kernel_launch(void* const* d_in, const int* in_sizes, int n_in,
                              void* d_out, int out_size, void* d_ws, size_t ws_size,
                              hipStream_t stream) {
    const float* x    = (const float*)d_in[0];
    const float* wu_i = (const float*)d_in[1];
    const float* wu_h = (const float*)d_in[2];
    const float* bu   = (const float*)d_in[3];
    const float* wr_i = (const float*)d_in[4];
    const float* wr_h = (const float*)d_in[5];
    const float* br   = (const float*)d_in[6];
    const float* wc_i = (const float*)d_in[7];
    const float* wc_h = (const float*)d_in[8];
    const float* bc   = (const float*)d_in[9];
    float* out = (float*)d_out;

    // Workspace layout:
    //   [0)              6x 512*512 bf16 transposed weights   (3 MB)
    //   [+3 MB)          bf16 copy of x, row-major (BT x 512) (64 MB)
    //   [+67 MB)         fp32 u_x (BT x 512)                  (128 MB)
    //   [+195 MB)        fp32 r_x (BT x 512)                  (128 MB)
    const size_t WSZ = (size_t)GRU_H * GRU_H;            // 262144 elems
    __bf16* wT    = (__bf16*)d_ws;
    __bf16* wT_ui = wT + 0 * WSZ;
    __bf16* wT_ri = wT + 1 * WSZ;
    __bf16* wT_ci = wT + 2 * WSZ;
    __bf16* wT_uh = wT + 3 * WSZ;
    __bf16* wT_rh = wT + 4 * WSZ;
    __bf16* wT_ch = wT + 5 * WSZ;
    __bf16* xb    = wT + 6 * WSZ;
    float*  u_x   = (float*)(xb + (size_t)GRU_BT * GRU_H);
    float*  r_x   = u_x + (size_t)GRU_BT * GRU_H;

    // Phase 0a: 6 weight transposes to bf16 (wT[n][k]); tiny (3 MB total).
    const int tgrid = (GRU_H * GRU_H) / 256;             // 1024 blocks
    wt_convert<<<tgrid, 256, 0, stream>>>(wu_i, wT_ui);
    wt_convert<<<tgrid, 256, 0, stream>>>(wr_i, wT_ri);
    wt_convert<<<tgrid, 256, 0, stream>>>(wc_i, wT_ci);
    wt_convert<<<tgrid, 256, 0, stream>>>(wu_h, wT_uh);
    wt_convert<<<tgrid, 256, 0, stream>>>(wr_h, wT_rh);
    wt_convert<<<tgrid, 256, 0, stream>>>(wc_h, wT_ch);

    // Phase 0b: x -> bf16, 8 elems/thread.
    const int xgrid = (int)(((size_t)GRU_BT * GRU_H) / (256 * 8));   // 16384 blocks
    x_convert<<<xgrid, 256, 0, stream>>>(x, xb);

    // Phase 1: fused projections. 4096 m-tiles x 32 n-tiles, 8 waves/block.
    proj_kernel<<<(4096 * 32) / 8, 256, 0, stream>>>(
        xb, wT_ui, wT_ri, wT_ci, bu, br, bc, u_x, r_x, out);

    // Phase 2: persistent scan, 2 WGs (16 batches each), no cross-WG sync.
    scan_kernel<<<2, 512, 0, stream>>>(wT_uh, wT_rh, wT_ch, u_x, r_x, out);
}
